// PermutationInvariantNet_26774826123434
// MI455X (gfx1250) — matrix operand order repaired and checked by reading, not compiled, not run on hardware
//
#include <hip/hip_runtime.h>
#include <hip/hip_bf16.h>

#define NSET   256   // nodes per set (= blockDim)
#define DM     128   // model dim D
#define HD     512   // hidden dim H
#define NLAYER 2
#define AP     136   // padded half-stride for 128-wide K buffers (bank-conflict free)
#define PP     72    // padded half-stride for 64-wide relu staging

// ---- staged-weight workspace layout (f16, matches LDS tile layouts exactly) ----
#define WP_TILE  (64 * AP)            // 8704 halves : Wp block  [n=64][k=128 pad AP]
#define W1_TILE  (64 * AP)            // 8704 halves : W1 block  [n=64][k=128 pad AP]
#define W2_TILE  (128 * PP)           // 9216 halves : W2 block  [n=128][k=64 pad PP]
#define WP_OFF   0
#define W1_OFF   (4 * WP_TILE)        // 34816
#define W2_OFF   (W1_OFF + 8 * W1_TILE)   // 104448
#define LAYER_WS (W2_OFF + 8 * W2_TILE)   // 178176 halves per layer
#define N8_WPW1  ((WP_TILE * 2) / 8)  // 2176 eight-byte units per Wp/W1 tile
#define N8_W2    ((W2_TILE * 2) / 8)  // 2304 eight-byte units per W2 tile

typedef __attribute__((ext_vector_type(16))) _Float16 v16h;
typedef __attribute__((ext_vector_type(8)))  _Float16 v8h;
typedef __attribute__((ext_vector_type(8)))  float    v8f;
typedef __attribute__((ext_vector_type(4)))  float    v4f;
typedef __attribute__((ext_vector_type(4)))  unsigned int u32x4;
typedef __attribute__((ext_vector_type(8)))  int      i32x8;
typedef __attribute__((ext_vector_type(4)))  int      i32x4;

#if __has_builtin(__builtin_amdgcn_tensor_load_to_lds) && __has_builtin(__builtin_amdgcn_s_wait_tensorcnt)
#define HAVE_TDM 1
#else
#define HAVE_TDM 0
#endif

// One-shot TDM DMA of a flat f16 tile (n8 units of 8 bytes) from global into LDS,
// then drain TENSORcnt. Descriptor built per CDNA5 ISA D# layout (CDNA5 ISA).
__device__ __forceinline__ void tdm_copy_wait(void* lds_dst, const void* gsrc, unsigned n8) {
#if HAVE_TDM
  const unsigned lds_addr = (unsigned)(unsigned long long)lds_dst;   // low 32b of flat = LDS byte offset
  const unsigned long long ga = (unsigned long long)gsrc;
  u32x4 g0;
  g0[0] = 1u;                                            // count=1 (valid user descriptor)
  g0[1] = lds_addr;                                      // lds_addr [63:32]
  g0[2] = (unsigned)(ga & 0xFFFFFFFFu);                  // global_addr [95:64]
  g0[3] = ((unsigned)(ga >> 32) & 0x01FFFFFFu)           // global_addr [120:96]
          | 0x80000000u;                                 // type=2 ("image") [127:126]
  i32x8 g1;
  g1[0] = (int)(3u << 16);                               // wg_mask=0, data_size=3 (8B units)
  g1[1] = (int)((n8 & 0xFFFFu) << 16);                   // tensor_dim0[15:0] @ bits 63:48
  g1[2] = (int)(((n8 >> 16) & 0xFFFFu) | (1u << 16));    // tensor_dim0[31:16] | tensor_dim1=1
  g1[3] = (int)((n8 & 0xFFFFu) << 16);                   // tile_dim0 = n8 @ bits 127:112
  g1[4] = 0;                                             // tile_dim1=0, tile_dim2=0 (1-D tile)
  g1[5] = (int)n8;                                       // tensor_dim0_stride[31:0]
  g1[6] = 0;
  g1[7] = 0;
  i32x4 z4 = {0, 0, 0, 0};
#if defined(__clang_major__) && (__clang_major__ >= 23)
  i32x8 z8 = {0, 0, 0, 0, 0, 0, 0, 0};
  __builtin_amdgcn_tensor_load_to_lds(g0, g1, z4, z4, z8, 0);
#else
  __builtin_amdgcn_tensor_load_to_lds(g0, g1, z4, z4, 0);
#endif
  __builtin_amdgcn_s_wait_tensorcnt((short)0);
#else
  (void)lds_dst; (void)gsrc; (void)n8;
#endif
}

// ---- WMMA operand loaders (CDNA5 16x16x32 f16 layouts, ISA 7.12.2) ----

__device__ __forceinline__ v16h load_a16(const _Float16* __restrict__ base, int stride,
                                         int row0, int k0, int lane) {
  const int m  = lane & 15;
  const int ko = (lane >> 4) << 3;            // 0 or 8
  const _Float16* p = base + (row0 + m) * stride + k0 + ko;
  v8h lo = *(const v8h*)(p);
  v8h hi = *(const v8h*)(p + 16);
  v16h a;
#pragma unroll
  for (int i = 0; i < 8; ++i) { a[i] = lo[i]; a[i + 8] = hi[i]; }
  return a;
}

__device__ __forceinline__ v16h load_b16(const _Float16* __restrict__ wt, int stride,
                                         int n0, int k0, int lane) {
  const int n  = lane & 15;
  const int kg = (lane >> 4) << 3;            // 0 or 8
  const _Float16* p = wt + (n0 + n) * stride + k0 + kg;
  v8h lo = *(const v8h*)(p);
  v8h hi = *(const v8h*)(p + 16);
  v16h b;
#pragma unroll
  for (int i = 0; i < 8; ++i) { b[i] = lo[i]; b[i + 8] = hi[i]; }
  return b;
}

// Synthesized A-tile of the leave-one-out pooled matrix
__device__ __forceinline__ v16h load_a_pooled(const float* __restrict__ cm1,
                                              const float* __restrict__ cm2,
                                              const int* __restrict__ cix,
                                              int row0, int k0, int lane) {
  const int row = row0 + (lane & 15);
  const int ko  = (lane >> 4) << 3;
  v16h a;
#pragma unroll
  for (int i = 0; i < 8; ++i) {
    const int k1 = k0 + ko + i;
    const int k2 = k1 + 16;
    a[i]     = (_Float16)((row == cix[k1]) ? cm2[k1] : cm1[k1]);
    a[i + 8] = (_Float16)((row == cix[k2]) ? cm2[k2] : cm1[k2]);
  }
  return a;
}

__device__ __forceinline__ v8f wmma16(v16h a, v16h b, v8f c) {
  return __builtin_amdgcn_wmma_f32_16x16x32_f16(false, a, false, b, (short)0, c, false, false);
}

// Row LayerNorm over 128 features; updates fp32 master row + f16 shadow row
__device__ __forceinline__ void layernorm_row(float* __restrict__ hrow,
                                              _Float16* __restrict__ arow,
                                              const float* __restrict__ g,
                                              const float* __restrict__ bg) {
  float s = 0.0f, s2 = 0.0f;
#pragma unroll
  for (int j = 0; j < DM; j += 4) {
    v4f v = *(const v4f*)(hrow + j);
#pragma unroll
    for (int c = 0; c < 4; ++c) { s += v[c]; s2 += v[c] * v[c]; }
  }
  const float mean = s * (1.0f / DM);
  const float var  = s2 * (1.0f / DM) - mean * mean;
  const float rs   = rsqrtf(var + 1e-5f);
#pragma unroll
  for (int j = 0; j < DM; j += 8) {
    v4f v0 = *(const v4f*)(hrow + j);
    v4f v1 = *(const v4f*)(hrow + j + 4);
    v4f g0 = *(const v4f*)(g + j);
    v4f g1 = *(const v4f*)(g + j + 4);
    v4f b0 = *(const v4f*)(bg + j);
    v4f b1v = *(const v4f*)(bg + j + 4);
    v4f o0, o1; v8h h8;
#pragma unroll
    for (int c = 0; c < 4; ++c) {
      o0[c] = (v0[c] - mean) * rs * g0[c] + b0[c];
      o1[c] = (v1[c] - mean) * rs * g1[c] + b1v[c];
      h8[c]     = (_Float16)o0[c];
      h8[4 + c] = (_Float16)o1[c];
    }
    *(v4f*)(hrow + j)     = o0;
    *(v4f*)(hrow + j + 4) = o1;
    *(v8h*)(arow + j)     = h8;
  }
}

// Fallback staging: kdim x ndim block of row-major (K x N) fp32 weight -> LDS f16 [n][k]
__device__ __forceinline__ void stage_w(const float* __restrict__ W, int ldn,
                                        int k_off, int n_off, int kdim, int ndim,
                                        _Float16* __restrict__ dst, int dstride, int tid) {
  const int total = kdim * ndim;
  for (int idx = tid; idx < total; idx += 256) {
    const int k = idx / ndim;
    const int n = idx - k * ndim;
    dst[n * dstride + k] = (_Float16)W[(size_t)(k_off + k) * ldn + n_off + n];
  }
}

// ---- prep kernels: convert fp32 weights to f16 tiles in ws (TDM-ready layout) ----

__global__ void prep_wp(const float* __restrict__ Wp, _Float16* __restrict__ ws) {
  const int idx = blockIdx.x * 256 + threadIdx.x;
  if (idx >= NLAYER * 4 * WP_TILE) return;
  const int layer = idx / (4 * WP_TILE);
  int rem = idx - layer * 4 * WP_TILE;
  const int tile = rem / WP_TILE;
  const int e = rem - tile * WP_TILE;
  const int n = e / AP, k = e - n * AP;
  const int kc = tile >> 1, nh = tile & 1;
  float v = 0.0f;
  if (k < DM) v = Wp[((size_t)layer * 2 * DM + kc * DM + k) * DM + nh * 64 + n];
  ws[(size_t)layer * LAYER_WS + WP_OFF + tile * WP_TILE + e] = (_Float16)v;
}

__global__ void prep_w1(const float* __restrict__ W1, _Float16* __restrict__ ws) {
  const int idx = blockIdx.x * 256 + threadIdx.x;
  if (idx >= NLAYER * 8 * W1_TILE) return;
  const int layer = idx / (8 * W1_TILE);
  int rem = idx - layer * 8 * W1_TILE;
  const int hc = rem / W1_TILE;
  const int e = rem - hc * W1_TILE;
  const int n = e / AP, k = e - n * AP;
  float v = 0.0f;
  if (k < DM) v = W1[((size_t)layer * DM + k) * HD + hc * 64 + n];
  ws[(size_t)layer * LAYER_WS + W1_OFF + hc * W1_TILE + e] = (_Float16)v;
}

__global__ void prep_w2(const float* __restrict__ W2, _Float16* __restrict__ ws) {
  const int idx = blockIdx.x * 256 + threadIdx.x;
  if (idx >= NLAYER * 8 * W2_TILE) return;
  const int layer = idx / (8 * W2_TILE);
  int rem = idx - layer * 8 * W2_TILE;
  const int hc = rem / W2_TILE;
  const int e = rem - hc * W2_TILE;
  const int n = e / PP, kk = e - n * PP;
  float v = 0.0f;
  if (kk < 64) v = W2[((size_t)layer * HD + hc * 64 + kk) * DM + n];
  ws[(size_t)layer * LAYER_WS + W2_OFF + hc * W2_TILE + e] = (_Float16)v;
}

// ---- fused permutation-invariant network, one workgroup per set ----

__global__ __launch_bounds__(256, 1)
void pinet_fused(const float* __restrict__ x,
                 const float* __restrict__ We,  const float* __restrict__ be,
                 const float* __restrict__ ge,  const float* __restrict__ bge,
                 const float* __restrict__ Wp,  const float* __restrict__ bp,
                 const float* __restrict__ gp,  const float* __restrict__ bgp,
                 const float* __restrict__ W1,  const float* __restrict__ b1,
                 const float* __restrict__ W2,  const float* __restrict__ b2,
                 const float* __restrict__ gf,  const float* __restrict__ bgf,
                 const _Float16* __restrict__ wsT,   // staged f16 tiles (nullptr -> fallback)
                 float* __restrict__ out)
{
  __shared__ _Float16 lA[NSET * AP];          // 69632 B : h in f16 (WMMA A source)
  __shared__ _Float16 lP[8 * 32 * PP];        // 36864 B : per-wave relu staging
  __shared__ _Float16 lW[DM * AP];            // 34816 B : one transposed weight tile [n][k]
  __shared__ float cm1[DM];
  __shared__ float cm2[DM];
  __shared__ int   cix[DM];

  const int b    = blockIdx.x;
  const int tid  = threadIdx.x;
  const int lane = tid & 31;
  const int wid  = tid >> 5;
  const bool use_tdm = HAVE_TDM && (wsT != nullptr);

  float* __restrict__ hf = out + (size_t)b * NSET * DM;   // fp32 master h (lives in d_out)

  // ---- embedding: h = LN(x @ We + be), IN_DIM == 2 ----
  {
    const float* xb = x + (size_t)b * NSET * 2;
    const float x0 = xb[tid * 2 + 0];
    const float x1 = xb[tid * 2 + 1];
    float* hr = hf + tid * DM;
#pragma unroll 8
    for (int j = 0; j < DM; ++j)
      hr[j] = x0 * We[j] + x1 * We[DM + j] + be[j];
    layernorm_row(hr, lA + tid * AP, ge, bge);
  }
  __syncthreads();

  for (int layer = 0; layer < NLAYER; ++layer) {
    const float* WpL  = Wp  + (size_t)layer * 2 * DM * DM;
    const float* bpL  = bp  + layer * DM;
    const float* gpL  = gp  + layer * DM;
    const float* bgpL = bgp + layer * DM;
    const float* W1L  = W1  + (size_t)layer * DM * HD;
    const float* b1L  = b1  + layer * HD;
    const float* W2L  = W2  + (size_t)layer * HD * DM;
    const float* b2L  = b2  + layer * DM;
    const float* gfL  = gf  + layer * DM;
    const float* bgfL = bgf + layer * DM;
    const _Float16* wsL = wsT ? (wsT + (size_t)layer * LAYER_WS) : nullptr;

    // ---- per-feature top-2 + argmax over the set ----
    if (tid < DM) {
      float m1 = -3.402823466e38f, m2 = -3.402823466e38f;
      int i1 = 0;
      for (int r = 0; r < NSET; ++r) {
        const float v = (float)lA[r * AP + tid];
        if (v > m1) { m2 = m1; m1 = v; i1 = r; }
        else if (v > m2) { m2 = v; }
      }
      cm1[tid] = m1; cm2[tid] = m2; cix[tid] = i1;
    }
    __syncthreads();

    v8f acc[16];
#pragma unroll
    for (int t = 0; t < 16; ++t)
#pragma unroll
      for (int r = 0; r < 8; ++r) acc[t][r] = 0.0f;

    // ---- pool linear: h += [h | loo_pooled] @ Wp + bp ----
#pragma unroll
    for (int kc = 0; kc < 2; ++kc) {
#pragma unroll
      for (int nh = 0; nh < 2; ++nh) {
        __syncthreads();
        if (use_tdm) {
          if (wid == 0)
            tdm_copy_wait(lW, wsL + WP_OFF + (kc * 2 + nh) * WP_TILE, N8_WPW1);
        } else {
          stage_w(WpL, DM, kc * DM, nh * 64, DM, 64, lW, AP, tid);
        }
        __syncthreads();
#pragma unroll
        for (int mt = 0; mt < 2; ++mt) {
          const int row0 = (wid * 2 + mt) * 16;
#pragma unroll
          for (int ntl = 0; ntl < 4; ++ntl) {
            const int t = mt * 8 + nh * 4 + ntl;
#pragma unroll
            for (int kk = 0; kk < 4; ++kk) {
              v16h av = (kc == 0)
                  ? load_a16(lA, AP, row0, kk * 32, lane)
                  : load_a_pooled(cm1, cm2, cix, row0, kk * 32, lane);
              v16h bv = load_b16(lW, AP, ntl * 16, kk * 32, lane);
              acc[t] = wmma16(av, bv, acc[t]);
            }
          }
        }
      }
    }
    {
      const int cg = lane >> 4, nn = lane & 15;
#pragma unroll
      for (int mt = 0; mt < 2; ++mt) {
        const int row0 = (wid * 2 + mt) * 16;
#pragma unroll
        for (int nt = 0; nt < 8; ++nt) {
          const int col  = nt * 16 + nn;
          const float bs = bpL[col];
#pragma unroll
          for (int r = 0; r < 8; ++r)
            hf[(row0 + r + 8 * cg) * DM + col] += acc[mt * 8 + nt][r] + bs;
        }
      }
    }
    __syncthreads();
    layernorm_row(hf + tid * DM, lA + tid * AP, gpL, bgpL);
    __syncthreads();

    // ---- feed-forward: h += relu(h @ W1 + b1) @ W2 + b2 (8 H-chunks of 64, fused) ----
#pragma unroll
    for (int t = 0; t < 16; ++t)
#pragma unroll
      for (int r = 0; r < 8; ++r) acc[t][r] = 0.0f;

    for (int hc = 0; hc < 8; ++hc) {
      if (!use_tdm && hc < 7)
        __builtin_prefetch(&W1L[(size_t)(tid & 127) * HD + (hc + 1) * 64], 0, 1);
      __syncthreads();
      if (use_tdm) {
        if (wid == 0)
          tdm_copy_wait(lW, wsL + W1_OFF + hc * W1_TILE, N8_WPW1);
      } else {
        stage_w(W1L, HD, 0, hc * 64, DM, 64, lW, AP, tid);   // W1 block: K=128 x N=64
      }
      __syncthreads();

      // GEMM1 + bias + relu -> wave-private LDS slice (32 rows x 64 cols)
      _Float16* lps = lP + wid * 32 * PP;
#pragma unroll
      for (int mt = 0; mt < 2; ++mt) {
        const int row0 = (wid * 2 + mt) * 16;
#pragma unroll
        for (int ntl = 0; ntl < 4; ++ntl) {
          v8f c1;
#pragma unroll
          for (int r = 0; r < 8; ++r) c1[r] = 0.0f;
#pragma unroll
          for (int kk = 0; kk < 4; ++kk) {
            v16h av = load_a16(lA, AP, row0, kk * 32, lane);
            v16h bv = load_b16(lW, AP, ntl * 16, kk * 32, lane);
            c1 = wmma16(av, bv, c1);
          }
          const int cg = lane >> 4, nn = lane & 15;
          const int col = ntl * 16 + nn;
          const float bs = b1L[hc * 64 + col];
#pragma unroll
          for (int r = 0; r < 8; ++r) {
            float v = c1[r] + bs;
            v = v > 0.0f ? v : 0.0f;
            lps[(mt * 16 + r + 8 * cg) * PP + col] = (_Float16)v;
          }
        }
      }
      if (!use_tdm && hc < 7)
        __builtin_prefetch(&W2L[(size_t)((hc + 1) * 64 + (tid >> 1)) * DM], 0, 1);
      __syncthreads();
      if (use_tdm) {
        if (wid == 0)
          tdm_copy_wait(lW, wsL + W2_OFF + hc * W2_TILE, N8_W2);
      } else {
        stage_w(W2L, DM, hc * 64, 0, 64, DM, lW, PP, tid);   // W2 block: K=64 x N=128
      }
      __syncthreads();

      // GEMM2 partial: acc += relu_chunk @ W2_chunk (K=64, wave reads its own relu slice)
#pragma unroll
      for (int mt = 0; mt < 2; ++mt) {
#pragma unroll
        for (int nt = 0; nt < 8; ++nt) {
#pragma unroll
          for (int kk = 0; kk < 2; ++kk) {
            v16h av = load_a16(lps, PP, mt * 16, kk * 32, lane);
            v16h bv = load_b16(lW, PP, nt * 16, kk * 32, lane);
            acc[mt * 8 + nt] = wmma16(av, bv, acc[mt * 8 + nt]);
          }
        }
      }
    }
    {
      const int cg = lane >> 4, nn = lane & 15;
#pragma unroll
      for (int mt = 0; mt < 2; ++mt) {
        const int row0 = (wid * 2 + mt) * 16;
#pragma unroll
        for (int nt = 0; nt < 8; ++nt) {
          const int col  = nt * 16 + nn;
          const float bs = b2L[col];
#pragma unroll
          for (int r = 0; r < 8; ++r)
            hf[(row0 + r + 8 * cg) * DM + col] += acc[mt * 8 + nt][r] + bs;
        }
      }
    }
    __syncthreads();
    layernorm_row(hf + tid * DM, lA + tid * AP, gfL, bgfL);
    __syncthreads();
  }
}

extern "C" void kernel_launch(void* const* d_in, const int* in_sizes, int n_in,
                              void* d_out, int out_size, void* d_ws, size_t ws_size,
                              hipStream_t stream) {
  (void)in_sizes; (void)n_in; (void)out_size;
  const float* x   = (const float*)d_in[0];
  const float* We  = (const float*)d_in[1];
  const float* be  = (const float*)d_in[2];
  const float* ge  = (const float*)d_in[3];
  const float* bge = (const float*)d_in[4];
  const float* Wp  = (const float*)d_in[5];
  const float* bp  = (const float*)d_in[6];
  const float* gp  = (const float*)d_in[7];
  const float* bgp = (const float*)d_in[8];
  const float* W1  = (const float*)d_in[9];
  const float* b1  = (const float*)d_in[10];
  const float* W2  = (const float*)d_in[11];
  const float* b2  = (const float*)d_in[12];
  const float* gf  = (const float*)d_in[13];
  const float* bgf = (const float*)d_in[14];

  const size_t ws_need = (size_t)NLAYER * LAYER_WS * 2;   // bytes of f16 staged weights
  _Float16* wsT = nullptr;
  if (d_ws != nullptr && ws_size >= ws_need) {
    wsT = (_Float16*)d_ws;
    {
      const int tot = NLAYER * 4 * WP_TILE;
      prep_wp<<<dim3((tot + 255) / 256), dim3(256), 0, stream>>>(Wp, wsT);
    }
    {
      const int tot = NLAYER * 8 * W1_TILE;
      prep_w1<<<dim3((tot + 255) / 256), dim3(256), 0, stream>>>(W1, wsT);
    }
    {
      const int tot = NLAYER * 8 * W2_TILE;
      prep_w2<<<dim3((tot + 255) / 256), dim3(256), 0, stream>>>(W2, wsT);
    }
  }

  pinet_fused<<<dim3(8), dim3(256), 0, stream>>>(
      x, We, be, ge, bge, Wp, bp, gp, bgp, W1, b1, W2, b2, gf, bgf, wsT, (float*)d_out);
}
